// LSTMDeformer_24197845745837
// MI455X (gfx1250) — matrix operand order repaired and checked by reading, not compiled, not run on hardware
//
#include <hip/hip_runtime.h>

#define LATENT 128
#define HIDDEN 128
#define STEPS  4
#define NGATES 512   // 4*HIDDEN

typedef _Float16 v16h __attribute__((ext_vector_type(16)));
typedef __fp16   pk2_t __attribute__((ext_vector_type(2)));   // return type of cvt_pkrtz
typedef float    v8f  __attribute__((ext_vector_type(8)));

// ---------- fast transcendentals ----------
__device__ __forceinline__ float fast_tanh(float x) {
#if __has_builtin(__builtin_amdgcn_tanhf)
    return __builtin_amdgcn_tanhf(x);
#else
    float t = __expf(-2.0f * fabsf(x));
    float r = (1.0f - t) / (1.0f + t);
    return copysignf(r, x);
#endif
}
__device__ __forceinline__ float fast_sigmoid(float x) {
    return 0.5f * fast_tanh(0.5f * x) + 0.5f;
}

// ---------------------------------------------------------------------------
// Pack kernel 1: W_hh (512x128 f32) -> f16 A-matrix fragments.
// A layout (16-bit A, 16x32 per fragment), contiguous k order used here:
//   lanes 0-15 : M=lane,    K = 0..15
//   lanes 16-31: M=lane-16, K = 16..31
// packed element j (0..15) of lane:  K = kc*32 + (lane/16)*16 + j
// Buffer: [32 mt][4 kc][32 lane][16 f16] = 128 KB.
// ---------------------------------------------------------------------------
__global__ void pack_whh_kernel(const float* __restrict__ W_hh,
                                _Float16* __restrict__ packA) {
    int tid  = blockIdx.x * 256 + threadIdx.x;          // 65536 threads
    int j    = tid & 15;
    int lane = (tid >> 4) & 31;
    int kc   = (tid >> 9) & 3;
    int mt   = tid >> 11;
    int f    = mt * 16 + (lane & 15);
    int K    = kc * 32 + ((lane >> 4) << 4) + j;
    packA[tid] = (_Float16)W_hh[f * HIDDEN + K];
}

// Pack kernel 2: W_a (6x128) zero-padded to 16x128, same A layout. 4 KB.
__global__ void pack_wa_kernel(const float* __restrict__ W_a,
                               _Float16* __restrict__ packA) {
    int tid  = blockIdx.x * 256 + threadIdx.x;          // 2048 threads
    int j    = tid & 15;
    int lane = (tid >> 4) & 31;
    int kc   = tid >> 9;
    int M    = lane & 15;
    int K    = kc * 32 + ((lane >> 4) << 4) + j;
    packA[tid] = (M < 6) ? (_Float16)W_a[M * HIDDEN + K] : (_Float16)0.0f;
}

// Pack kernel 3: per-batch per-gate-feature constants:
//   pack[b][f] = {b_ih[f]+b_hh[f]+dot(W_ih[f][0:128],latent[b]),
//                 W_ih[f][128], W_ih[f][129], W_ih[f][130]}
__global__ void pack_bx_kernel(const float* __restrict__ lat,
                               const float* __restrict__ W_ih,
                               const float* __restrict__ b_ih,
                               const float* __restrict__ b_hh,
                               float4* __restrict__ packBX) {
    int tid = blockIdx.x * 256 + threadIdx.x;           // 4096 threads
    int b   = tid >> 9;
    int f   = tid & 511;
    const float* wrow = W_ih + f * (LATENT + 3);
    const float* lrow = lat + b * LATENT;
    float s = b_ih[f] + b_hh[f];
    for (int k = 0; k < LATENT; ++k) s += wrow[k] * lrow[k];
    float4 o; o.x = s; o.y = wrow[LATENT]; o.z = wrow[LATENT + 1]; o.w = wrow[LATENT + 2];
    packBX[tid] = o;
}

// ---------------------------------------------------------------------------
// Main kernel: 256 threads = 8 waves; each wave owns 16 points.
// ---------------------------------------------------------------------------
__global__ void __launch_bounds__(256)
lstm_deformer_kernel(const float* __restrict__ x,
                     const float* __restrict__ b_a,
                     const _Float16* __restrict__ packWhh,
                     const _Float16* __restrict__ packWa,
                     const float4* __restrict__ packBX,
                     float* __restrict__ out) {
    __shared__ __align__(16) float4    s_pack[NGATES];        // 8 KB (per batch)
    __shared__ __align__(32) _Float16  s_hB[8][16][HIDDEN];   // 32 KB (per-wave h, point-major)

    const int tidx  = threadIdx.x;
    const int wave  = tidx >> 5;
    const int lane  = tidx & 31;
    const int n     = lane & 15;       // point within tile
    const int half  = lane >> 4;       // C-layout M half
    const int batch = blockIdx.x >> 9; // 512 blocks per batch (65536/128)

    // stage per-batch {base, wx, wy, wz} table into LDS
    {
        const float4* src = packBX + batch * NGATES;
        for (int i = tidx; i < NGATES; i += 256) s_pack[i] = src[i];
    }
    __syncthreads();

    const size_t p = (size_t)blockIdx.x * 128 + (size_t)wave * 16 + n;
    float px = x[p * 3 + 0];
    float py = x[p * 3 + 1];
    float pz = x[p * 3 + 2];

    // W_a A-fragments: resident in registers for whole kernel (32 VGPRs)
    v16h aWa[4];
#pragma unroll
    for (int kc = 0; kc < 4; ++kc)
        aWa[kc] = *(const v16h*)(packWa + (kc * 32 + lane) * 16);

    // b_a in C layout: row j = v + 8*half (rows >=6 are padding)
    float ba6[8];
#pragma unroll
    for (int v = 0; v < 8; ++v)
        ba6[v] = (half == 0 && v < 6) ? b_a[v] : 0.0f;

    float cst[8][8];                   // cell state, C layout, 64 VGPRs
#pragma unroll
    for (int t = 0; t < 8; ++t)
#pragma unroll
        for (int v = 0; v < 8; ++v) cst[t][v] = 0.0f;

    v16h Bfrag[4];                     // h^T B fragments (f16), 32 VGPRs
#pragma unroll
    for (int kc = 0; kc < 4; ++kc) Bfrag[kc] = (v16h)(_Float16)0.0f;

    int zofs = 0;                      // laundered zero: defeats LICM across steps

#pragma unroll 1
    for (int step = 0; step < STEPS; ++step) {
        // Optimization barrier: compiler must assume zofs changes each
        // iteration, so weight/bias loads below cannot be hoisted out of the
        // step loop (which previously caused ~1.5K VGPRs of hoisted loads to
        // spill to scratch).
        asm volatile("" : "+v"(zofs));

        // ---- gates^T + LSTM elementwise, per hidden 16-tile ----
#pragma unroll
        for (int ht = 0; ht < 8; ++ht) {
            auto gate_tile = [&](int mt) -> v8f {
                v8f acc;
#pragma unroll
                for (int v = 0; v < 8; ++v) {
                    float4 pw = s_pack[zofs + mt * 16 + v + 8 * half];
                    acc[v] = pw.x + pw.y * px + pw.z * py + pw.w * pz;
                }
                if (step != 0) {       // h == 0 on first step
#pragma unroll
                    for (int kc = 0; kc < 4; ++kc) {
                        const v16h A = *(const v16h*)(packWhh +
                            (size_t)(zofs + ((mt * 4 + kc) * 32 + lane) * 16));
                        acc = __builtin_amdgcn_wmma_f32_16x16x32_f16(
                            false, A, false, Bfrag[kc], (short)0, acc, false, false);
                    }
                }
                return acc;
            };
            v8f gf = gate_tile(8 + ht);    // forget
            v8f gi = gate_tile(ht);        // input
            v8f gg = gate_tile(16 + ht);   // cell
            v8f go = gate_tile(24 + ht);   // output

            float hval[8];
#pragma unroll
            for (int v = 0; v < 8; ++v) {
                float cv = cst[ht][v];
                cv = fast_sigmoid(gf[v]) * cv + fast_sigmoid(gi[v]) * fast_tanh(gg[v]);
                cst[ht][v] = cv;
                hval[v] = fast_sigmoid(go[v]) * fast_tanh(cv);
            }
            // store h (f16, point-major) for B-fragment rebuild
            const int featBase = ht * 16 + 8 * half;
#pragma unroll
            for (int i2 = 0; i2 < 4; ++i2) {
                pk2_t ph = __builtin_amdgcn_cvt_pkrtz(hval[2 * i2], hval[2 * i2 + 1]);
                *(pk2_t*)&s_hB[wave][n][featBase + 2 * i2] = ph;
            }
        }

        // ---- rebuild B fragments from fresh h ----
        // B k order used here (matches A packing): lanes 0-15: N=lane, K=0..15; lanes 16-31: N=lane-16, K=16..31
#pragma unroll
        for (int kc = 0; kc < 4; ++kc)
            Bfrag[kc] = *(const v16h*)&s_hB[wave][n][kc * 32 + half * 16];

        // ---- a = W_a_ext @ h^T + b_a  (4 WMMAs) ----
        v8f aAcc;
#pragma unroll
        for (int v = 0; v < 8; ++v) aAcc[v] = ba6[v];
#pragma unroll
        for (int kc = 0; kc < 4; ++kc)
            aAcc = __builtin_amdgcn_wmma_f32_16x16x32_f16(
                false, aWa[kc], false, Bfrag[kc], (short)0, aAcc, false, false);

        // a[j] for point n lives in lane n (half 0), VGPR j — broadcast to all lanes
        float a0 = __shfl(aAcc[0], n, 32);
        float a1 = __shfl(aAcc[1], n, 32);
        float a2 = __shfl(aAcc[2], n, 32);
        float a3 = __shfl(aAcc[3], n, 32);
        float a4 = __shfl(aAcc[4], n, 32);
        float a5 = __shfl(aAcc[5], n, 32);

        px = a3 + (1.0f + a0) * px;
        py = a4 + (1.0f + a1) * py;
        pz = a5 + (1.0f + a2) * pz;
    }

    if (half == 0) {
        out[p * 3 + 0] = px;
        out[p * 3 + 1] = py;
        out[p * 3 + 2] = pz;
    }
}

// ---------------------------------------------------------------------------
extern "C" void kernel_launch(void* const* d_in, const int* in_sizes, int n_in,
                              void* d_out, int out_size, void* d_ws, size_t ws_size,
                              hipStream_t stream) {
    const float* x    = (const float*)d_in[0];   // (8,65536,3)
    const float* lat  = (const float*)d_in[1];   // (8,128)
    const float* W_ih = (const float*)d_in[2];   // (512,131)
    const float* W_hh = (const float*)d_in[3];   // (512,128)
    const float* b_ih = (const float*)d_in[4];   // (512)
    const float* b_hh = (const float*)d_in[5];   // (512)
    const float* W_a  = (const float*)d_in[6];   // (6,128)
    const float* b_a  = (const float*)d_in[7];   // (6)
    float* out = (float*)d_out;

    _Float16* packWhh = (_Float16*)d_ws;                              // 128 KB
    _Float16* packWa  = (_Float16*)((char*)d_ws + 131072);            //   4 KB
    float4*   packBX  = (float4*)((char*)d_ws + 135168);              //  64 KB

    pack_whh_kernel<<<256, 256, 0, stream>>>(W_hh, packWhh);
    pack_wa_kernel<<<8, 256, 0, stream>>>(W_a, packWa);
    pack_bx_kernel<<<16, 256, 0, stream>>>(lat, W_ih, b_ih, b_hh, packBX);

    // 524288 points / 128 per block = 4096 blocks
    lstm_deformer_kernel<<<4096, 256, 0, stream>>>(x, b_a, packWhh, packWa, packBX, out);
}
